// GAT_Graph_18081812316386
// MI455X (gfx1250) — matrix operand order repaired and checked by reading, not compiled, not run on hardware
//
#include <hip/hip_runtime.h>
#include <math.h>

// ---------------- problem constants (from reference) ----------------
#define N_NODES  20000
#define N_EDGES  320000
#define EE       (N_EDGES + N_NODES)   // edges + self-loops
#define IN_CH    256
#define HID      128
#define HEADS    4
#define OUT_CH   16
#define N_GRAPHS 128
#define NEG_SLOPE 0.2f

typedef __bf16 bf16_t;
typedef __attribute__((ext_vector_type(16))) __bf16 v16bf;
typedef __attribute__((ext_vector_type(8)))  __bf16 v8bf;
typedef __attribute__((ext_vector_type(8)))  float  v8f;

// f32 -> bf16 round-to-nearest-even (bit manipulation; no reliance on cast lowering)
__device__ __forceinline__ bf16_t f2bf(float f) {
    unsigned u = __float_as_uint(f);
    unsigned r = (u + 0x7FFFu + ((u >> 16) & 1u)) >> 16;
    unsigned short s = (unsigned short)r;
    return *reinterpret_cast<bf16_t*>(&s);
}

// order-preserving f32<->u32 mapping so atomicMax(uint) implements float max
__device__ __forceinline__ unsigned f32_ord(float f) {
    unsigned b = __float_as_uint(f);
    return (b & 0x80000000u) ? ~b : (b | 0x80000000u);
}
__device__ __forceinline__ float ord_f32(unsigned u) {
    return (u & 0x80000000u) ? __uint_as_float(u & 0x7FFFFFFFu)
                             : __uint_as_float(~u);
}
#define ORD_NEG_INF 0x007FFFFFu   // f32_ord(-inf)

// ---------------- utility kernels ----------------
__global__ void k_fill_u32(unsigned* __restrict__ p, unsigned v, int n) {
    int i = blockIdx.x * blockDim.x + threadIdx.x;
    if (i < n) p[i] = v;
}

__global__ void k_f32_to_bf16(const float* __restrict__ in, bf16_t* __restrict__ out, int n) {
    int i = blockIdx.x * blockDim.x + threadIdx.x;
    if (i < n) out[i] = f2bf(in[i]);
}

// W [K,N] row-major f32  ->  Wt [N,K] row-major bf16 (so WMMA B fragments load contiguously)
__global__ void k_w_to_bf16T(const float* __restrict__ W, bf16_t* __restrict__ Wt, int K, int N) {
    int i = blockIdx.x * blockDim.x + threadIdx.x;
    if (i < K * N) {
        int k = i / N, n = i % N;
        Wt[(size_t)n * K + k] = f2bf(W[i]);
    }
}

// ---------------- WMMA bf16 GEMM: C[M,N] = A[M,K] @ Bt[N,K]^T ----------------
// One wave -> one 16x16 tile; 8 waves/block cover a 16x128 strip.
// grid = (N/128, M/16), block = 256.  M%16==0, N%128==0 (512 or 128), K%32==0.
__global__ __launch_bounds__(256)
void k_gemm_bf16_wmma(const bf16_t* __restrict__ A, const bf16_t* __restrict__ Bt,
                      float* __restrict__ C, int M, int N, int K) {
    const int lane  = threadIdx.x & 31;
    const int wave  = threadIdx.x >> 5;
    const int nBase = (blockIdx.x * 8 + wave) * 16;
    const int mBase = blockIdx.y * 16;
    const int half  = lane >> 4;        // 0: lanes 0-15, 1: lanes 16-31
    const int l16   = lane & 15;

    // A fragment (16x32 bf16, ISA 7.12.2): lane half selects K sub-blocks 0-7/16-23 vs 8-15/24-31
    const bf16_t* aptr = A  + (size_t)(mBase + l16) * K + half * 8;
    // B fragment (32x16 bf16): lane = column, half selects K sub-blocks 0-7/16-23 vs 8-15/24-31 (from transposed W)
    const bf16_t* bptr = Bt + (size_t)(nBase + l16) * K + half * 8;

    v8f acc = {};
    for (int k0 = 0; k0 < K; k0 += 32) {
        if (k0 + 32 < K) {                       // gfx1250 global_prefetch_b8
            __builtin_prefetch(aptr + k0 + 32, 0, 0);
            __builtin_prefetch(bptr + k0 + 32, 0, 0);
        }
        v16bf a, b;
        *reinterpret_cast<v8bf*>(&a)           = *reinterpret_cast<const v8bf*>(aptr + k0);
        *(reinterpret_cast<v8bf*>(&a) + 1)     = *reinterpret_cast<const v8bf*>(aptr + k0 + 16);
        *reinterpret_cast<v8bf*>(&b)           = *reinterpret_cast<const v8bf*>(bptr + k0);
        *(reinterpret_cast<v8bf*>(&b) + 1)     = *reinterpret_cast<const v8bf*>(bptr + k0 + 16);
        acc = __builtin_amdgcn_wmma_f32_16x16x32_bf16(false, a, false, b,
                                                      (short)0, acc, false, false);
    }
    // C/D layout: VGPR r -> M = r (lanes 0-15) or 8+r (lanes 16-31); N = l16
    float* cptr = C + (size_t)(mBase + half * 8) * N + nBase + l16;
#pragma unroll
    for (int r = 0; r < 8; ++r) cptr[(size_t)r * N] = acc[r];
}

// ---------------- attention scores: asrc[n,h] = <h[n,h,:], att_src[h,:]> ----------------
__global__ void k_att_scores(const float* __restrict__ h, const float* __restrict__ as,
                             const float* __restrict__ ad, float* __restrict__ asrc,
                             float* __restrict__ adst, int Hh, int Cc) {
    int i = blockIdx.x * blockDim.x + threadIdx.x;
    if (i >= N_NODES * Hh) return;
    int n = i / Hh, hd = i % Hh;
    const float* hp = h + (size_t)n * Hh * Cc + (size_t)hd * Cc;
    const float* sp = as + (size_t)hd * Cc;
    const float* dp = ad + (size_t)hd * Cc;
    float s = 0.f, d = 0.f;
    for (int c = 0; c < Cc; ++c) { float v = hp[c]; s += v * sp[c]; d += v * dp[c]; }
    asrc[i] = s; adst[i] = d;
}

__device__ __forceinline__ void edge_sd(const int* __restrict__ ei, int e, int& s, int& d) {
    if (e < N_EDGES) { s = ei[e]; d = ei[N_EDGES + e]; }
    else             { s = d = e - N_EDGES; }          // self loop
}

// segment max of leaky_relu(asrc[src]+adst[dst]) over dst, via ordered-uint atomicMax
__global__ void k_edge_max(const int* __restrict__ ei, const float* __restrict__ asrc,
                           const float* __restrict__ adst, unsigned* __restrict__ emax, int Hh) {
    int i = blockIdx.x * blockDim.x + threadIdx.x;
    if (i >= EE * Hh) return;
    int e = i / Hh, hd = i % Hh, s, d;
    edge_sd(ei, e, s, d);
    float x = asrc[s * Hh + hd] + adst[d * Hh + hd];
    x = x > 0.f ? x : NEG_SLOPE * x;
    atomicMax(&emax[d * Hh + hd], f32_ord(x));
}

// denom[dst,h] = sum exp(e - emax)
__global__ void k_edge_denom(const int* __restrict__ ei, const float* __restrict__ asrc,
                             const float* __restrict__ adst, const unsigned* __restrict__ emax,
                             float* __restrict__ denom, int Hh) {
    int i = blockIdx.x * blockDim.x + threadIdx.x;
    if (i >= EE * Hh) return;
    int e = i / Hh, hd = i % Hh, s, d;
    edge_sd(ei, e, s, d);
    float x = asrc[s * Hh + hd] + adst[d * Hh + hd];
    x = x > 0.f ? x : NEG_SLOPE * x;
    atomicAdd(&denom[d * Hh + hd], __expf(x - ord_f32(emax[d * Hh + hd])));
}

// agg[dst,h,c] += exp(e - emax) * h[src,h,c]   (normalized later by denom)
__global__ void k_edge_agg(const int* __restrict__ ei, const float* __restrict__ asrc,
                           const float* __restrict__ adst, const unsigned* __restrict__ emax,
                           const float* __restrict__ h, float* __restrict__ agg,
                           int Hh, int Cc) {
    int e = blockIdx.x, s, d;
    edge_sd(ei, e, s, d);
    int HC = Hh * Cc;
    for (int c = threadIdx.x; c < HC; c += blockDim.x) {
        int hd = c / Cc;
        float x = asrc[s * Hh + hd] + adst[d * Hh + hd];
        x = x > 0.f ? x : NEG_SLOPE * x;
        float alpha = __expf(x - ord_f32(emax[d * Hh + hd]));
        atomicAdd(&agg[(size_t)d * HC + c], alpha * h[(size_t)s * HC + c]);
    }
}

// layer1 finalize: relu(agg/denom + b1) -> bf16 (input to GEMM2)
__global__ void k_final1(const float* __restrict__ agg, const float* __restrict__ denom,
                         const float* __restrict__ b1, bf16_t* __restrict__ out) {
    int i = blockIdx.x * blockDim.x + threadIdx.x;
    if (i >= N_NODES * HEADS * HID) return;
    int n = i / (HEADS * HID), c = i % (HEADS * HID), hd = c / HID;
    float v = agg[i] / denom[n * HEADS + hd] + b1[c];
    out[i] = f2bf(v > 0.f ? v : 0.f);
}

// layer2 finalize: agg/denom + b2 -> f32
__global__ void k_final2(const float* __restrict__ agg, const float* __restrict__ denom,
                         const float* __restrict__ b2, float* __restrict__ out) {
    int i = blockIdx.x * blockDim.x + threadIdx.x;
    if (i >= N_NODES * HID) return;
    int n = i / HID, c = i % HID;
    out[i] = agg[i] / denom[n] + b2[c];
}

// global mean pool (sums + counts)
__global__ void k_pool(const float* __restrict__ h, const int* __restrict__ batch,
                       float* __restrict__ sums, float* __restrict__ cnts) {
    int i = blockIdx.x * blockDim.x + threadIdx.x;
    if (i >= N_NODES * HID) return;
    int n = i / HID, c = i % HID;
    int g = batch[n];
    atomicAdd(&sums[g * HID + c], h[i]);
    if (c == 0) atomicAdd(&cnts[g], 1.0f);
}

__global__ void k_mlp1(const float* __restrict__ sums, const float* __restrict__ cnts,
                       const float* __restrict__ Wm1, const float* __restrict__ bm1,
                       float* __restrict__ z) {
    int g = blockIdx.x, j = threadIdx.x;            // block = HID
    float inv = 1.0f / fmaxf(cnts[g], 1.0f);
    float s = bm1[j];
    for (int i = 0; i < HID; ++i) s += sums[g * HID + i] * inv * Wm1[i * HID + j];
    z[g * HID + j] = fmaxf(s, 0.f);
}

__global__ void k_mlp2(const float* __restrict__ z, const float* __restrict__ Wm2,
                       const float* __restrict__ bm2, float* __restrict__ out) {
    int g = blockIdx.x, j = threadIdx.x;            // block = OUT_CH
    float s = bm2[j];
    for (int i = 0; i < HID; ++i) s += z[g * HID + i] * Wm2[i * OUT_CH + j];
    out[g * OUT_CH + j] = s;
}

// ---------------- host side ----------------
static inline int cdiv(int a, int b) { return (a + b - 1) / b; }

extern "C" void kernel_launch(void* const* d_in, const int* in_sizes, int n_in,
                              void* d_out, int out_size, void* d_ws, size_t ws_size,
                              hipStream_t stream) {
    const float* x     = (const float*)d_in[0];
    const int*   ei    = (const int*)  d_in[1];   // [2, E]
    const int*   batch = (const int*)  d_in[2];
    const float* W1  = (const float*)d_in[3];
    const float* as1 = (const float*)d_in[4];
    const float* ad1 = (const float*)d_in[5];
    const float* b1  = (const float*)d_in[6];
    const float* W2  = (const float*)d_in[7];
    const float* as2 = (const float*)d_in[8];
    const float* ad2 = (const float*)d_in[9];
    const float* b2  = (const float*)d_in[10];
    const float* Wm1 = (const float*)d_in[11];
    const float* bm1 = (const float*)d_in[12];
    const float* Wm2 = (const float*)d_in[13];
    const float* bm2 = (const float*)d_in[14];
    float* out = (float*)d_out;

    char* ws = (char*)d_ws;
    size_t o = 0;
    auto alloc = [&](size_t bytes) { size_t r = o; o += (bytes + 255) & ~(size_t)255; return r; };

    bf16_t* xb    = (bf16_t*)(ws + alloc((size_t)N_NODES * IN_CH * 2));        // also reused as f32 h2out
    bf16_t* Wt1   = (bf16_t*)(ws + alloc((size_t)512 * 256 * 2));
    bf16_t* Wt2   = (bf16_t*)(ws + alloc((size_t)128 * 512 * 2));
    float*  h1    = (float*) (ws + alloc((size_t)N_NODES * 512 * 4));          // gemm1 out, reused for gemm2 out
    float*  asrc  = (float*) (ws + alloc((size_t)N_NODES * HEADS * 4));
    float*  adst  = (float*) (ws + alloc((size_t)N_NODES * HEADS * 4));
    unsigned* emax= (unsigned*)(ws + alloc((size_t)N_NODES * HEADS * 4));
    float*  denom = (float*) (ws + alloc((size_t)N_NODES * HEADS * 4));
    float*  agg   = (float*) (ws + alloc((size_t)N_NODES * 512 * 4));          // reused for layer-2 agg
    bf16_t* hrelu = (bf16_t*)(ws + alloc((size_t)N_NODES * 512 * 2));
    float*  pooled= (float*) (ws + alloc((size_t)N_GRAPHS * HID * 4));
    float*  cnts  = (float*) (ws + alloc((size_t)N_GRAPHS * 4));
    float*  zbuf  = (float*) (ws + alloc((size_t)N_GRAPHS * HID * 4));
    float*  h2out = (float*)xb;   // N_NODES*HID*4 == N_NODES*IN_CH*2 bytes, xb dead by then

    const int B = 256;

    // ---- convert inputs to bf16 (W transposed for B-fragment loads) ----
    k_f32_to_bf16<<<cdiv(N_NODES * IN_CH, B), B, 0, stream>>>(x, xb, N_NODES * IN_CH);
    k_w_to_bf16T<<<cdiv(256 * 512, B), B, 0, stream>>>(W1, Wt1, 256, 512);
    k_w_to_bf16T<<<cdiv(512 * 128, B), B, 0, stream>>>(W2, Wt2, 512, 128);

    // ---- layer 1: h1 = x @ W1  (WMMA) ----
    k_gemm_bf16_wmma<<<dim3(512 / 128, N_NODES / 16), B, 0, stream>>>(xb, Wt1, h1, N_NODES, 512, 256);
    k_att_scores<<<cdiv(N_NODES * HEADS, B), B, 0, stream>>>(h1, as1, ad1, asrc, adst, HEADS, HID);
    k_fill_u32<<<cdiv(N_NODES * HEADS, B), B, 0, stream>>>(emax, ORD_NEG_INF, N_NODES * HEADS);
    k_fill_u32<<<cdiv(N_NODES * HEADS, B), B, 0, stream>>>((unsigned*)denom, 0u, N_NODES * HEADS);
    k_fill_u32<<<cdiv(N_NODES * 512, B), B, 0, stream>>>((unsigned*)agg, 0u, N_NODES * 512);
    k_edge_max<<<cdiv(EE * HEADS, B), B, 0, stream>>>(ei, asrc, adst, emax, HEADS);
    k_edge_denom<<<cdiv(EE * HEADS, B), B, 0, stream>>>(ei, asrc, adst, emax, denom, HEADS);
    k_edge_agg<<<EE, 256, 0, stream>>>(ei, asrc, adst, emax, h1, agg, HEADS, HID);
    k_final1<<<cdiv(N_NODES * 512, B), B, 0, stream>>>(agg, denom, b1, hrelu);

    // ---- layer 2: h2 = relu(h1') @ W2  (WMMA), heads = 1 ----
    k_gemm_bf16_wmma<<<dim3(1, N_NODES / 16), B, 0, stream>>>(hrelu, Wt2, h1, N_NODES, 128, 512);
    k_att_scores<<<cdiv(N_NODES, B), B, 0, stream>>>(h1, as2, ad2, asrc, adst, 1, HID);
    k_fill_u32<<<cdiv(N_NODES, B), B, 0, stream>>>(emax, ORD_NEG_INF, N_NODES);
    k_fill_u32<<<cdiv(N_NODES, B), B, 0, stream>>>((unsigned*)denom, 0u, N_NODES);
    k_fill_u32<<<cdiv(N_NODES * HID, B), B, 0, stream>>>((unsigned*)agg, 0u, N_NODES * HID);
    k_edge_max<<<cdiv(EE, B), B, 0, stream>>>(ei, asrc, adst, emax, 1);
    k_edge_denom<<<cdiv(EE, B), B, 0, stream>>>(ei, asrc, adst, emax, denom, 1);
    k_edge_agg<<<EE, 128, 0, stream>>>(ei, asrc, adst, emax, h1, agg, 1, HID);
    k_final2<<<cdiv(N_NODES * HID, B), B, 0, stream>>>(agg, denom, b2, h2out);

    // ---- mean pool + MLP head ----
    k_fill_u32<<<cdiv(N_GRAPHS * HID, B), B, 0, stream>>>((unsigned*)pooled, 0u, N_GRAPHS * HID);
    k_fill_u32<<<1, N_GRAPHS, 0, stream>>>((unsigned*)cnts, 0u, N_GRAPHS);
    k_pool<<<cdiv(N_NODES * HID, B), B, 0, stream>>>(h2out, batch, pooled, cnts);
    k_mlp1<<<N_GRAPHS, HID, 0, stream>>>(pooled, cnts, Wm1, bm1, zbuf);
    k_mlp2<<<N_GRAPHS, OUT_CH, 0, stream>>>(zbuf, Wm2, bm2, out);

    (void)in_sizes; (void)n_in; (void)out_size; (void)ws_size;
}
